// BitMHA_39393440039204
// MI455X (gfx1250) — 16-bit and 8-bit matrix operand order repaired and checked by reading, compiled, not run on hardware
//
#include <hip/hip_runtime.h>

typedef __attribute__((ext_vector_type(8)))  int      v8i;
typedef __attribute__((ext_vector_type(8)))  float    v8f;
typedef __attribute__((ext_vector_type(16))) _Float16 v16h;

constexpr int   B_    = 2;
constexpr int   N_    = 2048;
constexpr int   C_    = 2048;
constexpr int   H_    = 16;
constexpr int   D_    = 128;
constexpr int   M_    = B_ * N_;
constexpr int   OQKV  = 3 * C_;
constexpr float SCALE_ = 0.08838834764831845f;
constexpr float NEG_   = -9.0e15f;

__global__ void k_zero4(float* acc) {
  if (threadIdx.x < 4) acc[threadIdx.x] = 0.0f;
}

__global__ void k_absmean(const float* __restrict__ w, long count,
                          float* __restrict__ acc, int slot) {
  float s = 0.0f;
  for (long i = blockIdx.x * (long)blockDim.x + threadIdx.x; i < count;
       i += (long)gridDim.x * blockDim.x)
    s += fabsf(w[i]);
  #pragma unroll
  for (int m = 16; m; m >>= 1) s += __shfl_xor(s, m, 32);
  __shared__ float red[8];
  const int lane = threadIdx.x & 31, wv = threadIdx.x >> 5;
  if (lane == 0) red[wv] = s;
  __syncthreads();
  if (threadIdx.x == 0) {
    float t = 0.0f;
    for (int i = 0; i < (int)(blockDim.x >> 5); ++i) t += red[i];
    atomicAdd(&acc[slot], t);
  }
}

__global__ void k_wfinalize(const float* __restrict__ acc, float* __restrict__ sc) {
  const float m0 = fmaxf(acc[0] / (float)((long)OQKV * C_), 1e-5f);
  const float m1 = fmaxf(acc[1] / (float)((long)C_   * C_), 1e-5f);
  sc[0] = 1.0f / m0; sc[1] = m0;
  sc[2] = 1.0f / m1; sc[3] = m1;
}

__global__ void k_wpack(const float* __restrict__ w, unsigned* __restrict__ packed,
                        int O, int K, const float* __restrict__ sc, int slot) {
  const int idx = blockIdx.x * blockDim.x + threadIdx.x;
  const int total = (O * K) >> 2;
  if (idx >= total) return;
  const float s = sc[slot];
  const int dw     = idx & 7;
  const int lane   = (idx >> 3) & 31;
  const int tile   = idx >> 8;
  const int ktiles = K >> 6;
  const int kt = tile % ktiles, nt = tile / ktiles;
  const int n  = nt * 16 + (lane & 15);
  const int kb = kt * 64 + (dw >> 1) * 16 + ((lane & 16) ? 8 : 0) + (dw & 1) * 4;
  const float* row = w + (long)n * K + kb;
  unsigned r = 0;
  #pragma unroll
  for (int j = 0; j < 4; ++j) {
    float q = rintf(row[j] * s);
    q = fminf(fmaxf(q, -1.0f), 1.0f);
    r |= ((unsigned)(int)q & 0xffu) << (8 * j);
  }
  packed[idx] = r;
}

__global__ void k_actquant(const float* __restrict__ x, signed char* __restrict__ xq,
                           float* __restrict__ deq, int K) {
  const int row = blockIdx.x;
  const float* src = x + (long)row * K;
  float mx = 0.0f;
  for (int i = threadIdx.x; i < K; i += blockDim.x) mx = fmaxf(mx, fabsf(src[i]));
  #pragma unroll
  for (int m = 16; m; m >>= 1) mx = fmaxf(mx, __shfl_xor(mx, m, 32));
  __shared__ float red[8];
  const int lane = threadIdx.x & 31, wv = threadIdx.x >> 5;
  if (lane == 0) red[wv] = mx;
  __syncthreads();
  mx = red[0];
  for (int i = 1; i < (int)(blockDim.x >> 5); ++i) mx = fmaxf(mx, red[i]);
  mx = fmaxf(mx, 1e-5f);
  const float s = 127.0f / mx;
  if (threadIdx.x == 0) deq[row] = mx / 127.0f;
  signed char* dst = xq + (long)row * K;
  for (int i = threadIdx.x; i < K; i += blockDim.x) {
    float q = rintf(src[i] * s);
    q = fminf(fmaxf(q, -128.0f), 127.0f);
    dst[i] = (signed char)(int)q;
  }
}

template <int MODE>
__global__ __launch_bounds__(256)
void k_gemm_iu8(const signed char* __restrict__ aq, const float* __restrict__ adeq,
                const unsigned* __restrict__ bpk, const float* __restrict__ sc,
                _Float16* __restrict__ qf, _Float16* __restrict__ kf,
                _Float16* __restrict__ vT, float* __restrict__ out,
                int Ktot, int Ncols) {
  const int wave = threadIdx.x >> 5;
  const int lane = threadIdx.x & 31;
  const int ln   = lane & 15;
  const int hi8  = (lane & 16) ? 8 : 0;
  const int m0 = blockIdx.y * 128 + (wave >> 2) * 64;
  const int n0 = blockIdx.x * 256 + (wave & 3) * 64;
  const int ktiles = Ktot >> 6;
  const float wdeq = sc[MODE == 0 ? 1 : 3];

  v8i acc[4][4];
  #pragma unroll
  for (int i = 0; i < 4; ++i)
    #pragma unroll
    for (int j = 0; j < 4; ++j) acc[i][j] = {};

  for (int kt = 0; kt < ktiles; ++kt) {
    const int kbase = kt * 64 + hi8;
    v8i afrag[4];
    #pragma unroll
    for (int mt = 0; mt < 4; ++mt) {
      const signed char* ap = aq + (long)(m0 + mt * 16 + ln) * Ktot + kbase;
      union { v8i v; int2 d[4]; } u;
      u.d[0] = *(const int2*)(ap + 0);
      u.d[1] = *(const int2*)(ap + 16);
      u.d[2] = *(const int2*)(ap + 32);
      u.d[3] = *(const int2*)(ap + 48);
      afrag[mt] = u.v;
    }
    #pragma unroll
    for (int nt = 0; nt < 4; ++nt) {
      const long tile = (long)((n0 >> 4) + nt) * ktiles + kt;
      const v8i bfrag = *(const v8i*)(bpk + (tile * 32 + lane) * 8);
      #pragma unroll
      for (int mt = 0; mt < 4; ++mt)
        acc[mt][nt] = __builtin_amdgcn_wmma_i32_16x16x64_iu8(
            true, afrag[mt], true, bfrag, acc[mt][nt], false, false);
    }
    if (kt + 1 < ktiles) {
      __builtin_prefetch(aq + (long)(m0 + ln) * Ktot + kbase + 64, 0, 1);
      __builtin_prefetch(bpk + (((long)(n0 >> 4) * ktiles + kt + 1) * 32 + lane) * 8, 0, 1);
    }
  }

  #pragma unroll
  for (int mt = 0; mt < 4; ++mt) {
    #pragma unroll
    for (int nt = 0; nt < 4; ++nt) {
      const int col = n0 + nt * 16 + ln;
      #pragma unroll
      for (int r = 0; r < 8; ++r) {
        const int row = m0 + mt * 16 + hi8 + r;
        const float y = (float)acc[mt][nt][r] * adeq[row] * wdeq;
        if (MODE == 1) {
          out[(long)row * Ncols + col] = y;
        } else {
          const int which = col >> 11;
          const int rem = col & 2047;
          const int h = rem >> 7, d = rem & 127;
          const int b = row >> 11, n = row & 2047;
          const long bh = (long)b * H_ + h;
          if (which == 0)      qf[(bh * N_ + n) * D_ + d] = (_Float16)(y * SCALE_);
          else if (which == 1) kf[(bh * N_ + n) * D_ + d] = (_Float16)y;
          else                 vT[(bh * D_ + d) * N_ + n] = (_Float16)y;
        }
      }
    }
  }
}

__global__ __launch_bounds__(256)
void k_attention(const _Float16* __restrict__ qf, const _Float16* __restrict__ kf,
                 const _Float16* __restrict__ vT, float* __restrict__ attn_out) {
  const int wave = threadIdx.x >> 5;
  const int lane = threadIdx.x & 31;
  const int ln   = lane & 15;
  const int hi8  = (lane & 16) ? 8 : 0;
  const int koff = (lane & 16) ? 16 : 0;
  const int bh = blockIdx.y;
  const int b = bh >> 4, h = bh & 15;
  const int q0 = blockIdx.x * 128 + wave * 16;

  __shared__ _Float16 plds[8][16 * 32];

  v16h qfr[4];
  {
    const _Float16* qrow = qf + ((long)bh * N_ + (q0 + ln)) * D_;
    #pragma unroll
    for (int c = 0; c < 4; ++c) {
      union { v16h hh; int4 i[2]; } u;
      u.i[0] = *(const int4*)(qrow + c * 32 + koff);
      u.i[1] = *(const int4*)(qrow + c * 32 + koff + 8);
      qfr[c] = u.hh;
    }
  }

  v8f o[8];
  #pragma unroll
  for (int t = 0; t < 8; ++t) o[t] = {};
  float mrun[8], lrun[8];
  #pragma unroll
  for (int r = 0; r < 8; ++r) { mrun[r] = -3.0e38f; lrun[r] = 0.0f; }

  const int jend = q0 + 16;
  for (int j = 0; j < jend; j += 32) {
    v8f s_lo = {}, s_hi = {};
    #pragma unroll
    for (int c = 0; c < 4; ++c) {
      const _Float16* klo = kf + ((long)bh * N_ + (j + ln)) * D_ + c * 32 + koff;
      const _Float16* khi = kf + ((long)bh * N_ + (j + 16 + ln)) * D_ + c * 32 + koff;
      const v16h kblo = *(const v16h*)klo;
      const v16h kbhi = *(const v16h*)khi;
      s_lo = __builtin_amdgcn_wmma_f32_16x16x32_f16(false, qfr[c], false, kblo,
                                                    (short)0, s_lo, false, false);
      s_hi = __builtin_amdgcn_wmma_f32_16x16x32_f16(false, qfr[c], false, kbhi,
                                                    (short)0, s_hi, false, false);
    }
    #pragma unroll
    for (int r = 0; r < 8; ++r) {
      const int row = q0 + hi8 + r;
      const int col_lo = j + ln;
      float slo = (col_lo      <= row) ? s_lo[r] : NEG_;
      float shi = (col_lo + 16 <= row) ? s_hi[r] : NEG_;
      float mx = fmaxf(slo, shi);
      #pragma unroll
      for (int msk = 8; msk; msk >>= 1) mx = fmaxf(mx, __shfl_xor(mx, msk, 32));
      const float mnew = fmaxf(mrun[r], mx);
      const float f = __expf(mrun[r] - mnew);
      const float plo = __expf(slo - mnew);
      const float phi = __expf(shi - mnew);
      float psum = plo + phi;
      #pragma unroll
      for (int msk = 8; msk; msk >>= 1) psum += __shfl_xor(psum, msk, 32);
      lrun[r] = lrun[r] * f + psum;
      mrun[r] = mnew;
      #pragma unroll
      for (int t = 0; t < 8; ++t) o[t][r] *= f;
      plds[wave][(hi8 + r) * 32 + ln]      = (_Float16)plo;
      plds[wave][(hi8 + r) * 32 + ln + 16] = (_Float16)phi;
    }
    asm volatile("s_wait_dscnt 0" ::: "memory");
    v16h pfr;
    {
      union { v16h hh; int4 i[2]; } u;
      const _Float16* pp = &plds[wave][ln * 32 + koff];
      u.i[0] = *(const int4*)(pp);
      u.i[1] = *(const int4*)(pp + 8);
      pfr = u.hh;
    }
    #pragma unroll
    for (int t = 0; t < 8; ++t) {
      const _Float16* vp = vT + ((long)bh * D_ + t * 16 + ln) * N_ + j + koff;
      const v16h vfr = *(const v16h*)vp;
      o[t] = __builtin_amdgcn_wmma_f32_16x16x32_f16(false, pfr, false, vfr,
                                                    (short)0, o[t], false, false);
    }
  }

  #pragma unroll
  for (int t = 0; t < 8; ++t) {
    #pragma unroll
    for (int r = 0; r < 8; ++r) {
      const int row = q0 + hi8 + r;
      const float y = o[t][r] / lrun[r];
      attn_out[((long)b * N_ + row) * C_ + h * D_ + t * 16 + ln] = y;
    }
  }
}

extern "C" void kernel_launch(void* const* d_in, const int* in_sizes, int n_in,
                              void* d_out, int out_size, void* d_ws, size_t ws_size,
                              hipStream_t stream) {
  (void)in_sizes; (void)n_in; (void)out_size; (void)ws_size;
  const float* x      = (const float*)d_in[0];
  const float* w_qkv  = (const float*)d_in[1];
  const float* w_proj = (const float*)d_in[2];
  float* out = (float*)d_out;

  char* ws = (char*)d_ws;
  size_t off = 0;
  auto take = [&](size_t bytes) -> void* {
    void* p = ws + off;
    off = (off + bytes + 255) & ~(size_t)255;
    return p;
  };
  float*       accum    = (float*)take(4 * sizeof(float));
  float*       sc       = (float*)take(4 * sizeof(float));
  signed char* xq       = (signed char*)take((size_t)M_ * C_);
  float*       xdeq     = (float*)take((size_t)M_ * sizeof(float));
  unsigned*    wqkv_pk  = (unsigned*)take((size_t)OQKV * C_);
  unsigned*    wproj_pk = (unsigned*)take((size_t)C_ * C_);
  _Float16*    qf       = (_Float16*)take((size_t)B_ * H_ * N_ * D_ * 2);
  _Float16*    kf       = (_Float16*)take((size_t)B_ * H_ * N_ * D_ * 2);
  _Float16*    vT       = (_Float16*)take((size_t)B_ * H_ * D_ * N_ * 2);
  float*       attn     = (float*)take((size_t)M_ * C_ * sizeof(float));
  signed char* oq       = (signed char*)take((size_t)M_ * C_);
  float*       odeq     = (float*)take((size_t)M_ * sizeof(float));

  k_zero4<<<1, 32, 0, stream>>>(accum);
  k_absmean<<<1024, 256, 0, stream>>>(w_qkv, (long)OQKV * C_, accum, 0);
  k_absmean<<<1024, 256, 0, stream>>>(w_proj, (long)C_ * C_, accum, 1);
  k_wfinalize<<<1, 1, 0, stream>>>(accum, sc);
  k_wpack<<<(OQKV * C_ / 4 + 255) / 256, 256, 0, stream>>>(w_qkv, wqkv_pk, OQKV, C_, sc, 0);
  k_wpack<<<(C_ * C_ / 4 + 255) / 256, 256, 0, stream>>>(w_proj, wproj_pk, C_, C_, sc, 2);

  k_actquant<<<M_, 256, 0, stream>>>(x, xq, xdeq, C_);
  k_gemm_iu8<0><<<dim3(OQKV / 256, M_ / 128), 256, 0, stream>>>(
      xq, xdeq, wqkv_pk, sc, qf, kf, vT, nullptr, C_, OQKV);

  k_attention<<<dim3(N_ / 128, B_ * H_), 256, 0, stream>>>(qf, kf, vT, attn);

  k_actquant<<<M_, 256, 0, stream>>>(attn, oq, odeq, C_);
  k_gemm_iu8<1><<<dim3(C_ / 256, M_ / 128), 256, 0, stream>>>(
      oq, odeq, wproj_pk, sc, nullptr, nullptr, nullptr, out, C_, C_);
}
